// LinearTemporalAttention_73443940761983
// MI455X (gfx1250) — matrix operand order repaired and checked by reading, not compiled, not run on hardware
//
#include <hip/hip_runtime.h>
#include <hip/hip_bf16.h>

#define C_DIM 1024
#define HEADS 16
#define D_DIM 64
#define BATCH 4
#define SEQ   4096
#define BT    (BATCH * SEQ)

#define BM 128
#define BN 128
#define BK 32

#define LDA  40    // 32 + 8 pad (ushorts), row = 80B (16B multiple)
#define LDB  72    // 64 + 8 pad (ushorts), row = 144B
#define LDB2 136   // 128 + 8 pad (ushorts), row = 272B
#define LDC  132   // 128 + 4 pad (floats)

#define KVSEG 8
#define TSEG  (SEQ / KVSEG)   // 512

typedef __bf16 bf16x16 __attribute__((ext_vector_type(16)));
typedef float  floatx8 __attribute__((ext_vector_type(8)));

union BFrag { bf16x16 v; unsigned short s[16]; };

__device__ __forceinline__ unsigned short f2bf(float f) {
    unsigned int u = __float_as_uint(f);
    u += 0x7FFFu + ((u >> 16) & 1u);      // round-to-nearest-even
    return (unsigned short)(u >> 16);
}
__device__ __forceinline__ float bf2f(unsigned short h) {
    return __uint_as_float(((unsigned int)h) << 16);
}

// ---- fragment gathers per CDNA5 ISA 7.12.2 (wave32) -------------------------
// A (16x32 bf16): lane = M (0..15, both halves), element e -> K = (e<8?e:e+8)+8*half
template <int LD>
__device__ __forceinline__ void load_a_frag_t(const unsigned short (*As)[LD],
                                              int mbase, int lane, BFrag& a) {
    const int m = mbase + (lane & 15);
    const int half = lane >> 4;
#pragma unroll
    for (int e = 0; e < 16; ++e) {
        int k = (e < 8 ? e : e + 8) + 8 * half;
        a.s[e] = As[m][k];
    }
}
// B (32x16 bf16): lane = N (0..15, both halves), element e -> K = (e<8?e:e+8)+8*half
template <int LD>
__device__ __forceinline__ void load_b_frag_t(const unsigned short (*Bs)[LD],
                                              int nbase, int lane, BFrag& b) {
#pragma unroll
    for (int e = 0; e < 16; ++e) b.s[e] = Bs[(e < 8 ? e : e + 8) + 8 * (lane >> 4)][nbase + (lane & 15)];
}

__device__ __forceinline__ floatx8 wmma_bf16(const BFrag& a, const BFrag& b, floatx8 c) {
    return __builtin_amdgcn_wmma_f32_16x16x32_bf16(false, a.v, false, b.v,
                                                   (short)0, c, false, false);
}

// ---- transpose x [B,C,T] f32 -> xt [B*T, C] bf16 ----------------------------
__global__ void transpose_x_kernel(const float* __restrict__ x,
                                   unsigned short* __restrict__ xt) {
    __shared__ float tile[32][33];
    const int b  = blockIdx.z;
    const int t0 = blockIdx.x * 32;
    const int c0 = blockIdx.y * 32;
    const int tx = threadIdx.x, ty = threadIdx.y;
#pragma unroll
    for (int i = 0; i < 4; ++i)
        tile[ty + i * 8][tx] = x[((size_t)b * C_DIM + c0 + ty + i * 8) * SEQ + t0 + tx];
    __syncthreads();
#pragma unroll
    for (int i = 0; i < 4; ++i)
        xt[((size_t)b * SEQ + t0 + ty + i * 8) * C_DIM + c0 + tx] =
            f2bf(tile[tx][ty + i * 8]);
}

// ---- Wt[c][j] = W[j][c], f32 -> bf16 ---------------------------------------
__global__ void transpose_w_kernel(const float* __restrict__ W,
                                   unsigned short* __restrict__ Wt) {
    __shared__ float tile[32][33];
    const int j0 = blockIdx.x * 32;
    const int c0 = blockIdx.y * 32;
    const int tx = threadIdx.x, ty = threadIdx.y;
#pragma unroll
    for (int i = 0; i < 4; ++i)
        tile[ty + i * 8][tx] = W[(size_t)(j0 + ty + i * 8) * C_DIM + c0 + tx];
    __syncthreads();
#pragma unroll
    for (int i = 0; i < 4; ++i)
        Wt[(size_t)(c0 + ty + i * 8) * C_DIM + j0 + tx] = f2bf(tile[tx][ty + i * 8]);
}

// ---- QKV projection: out = act(xt @ W^T + b) --------------------------------
// 128x128 tile, 8 waves, each wave 32x64, double-buffered LDS, reg-staged loads
__global__ void __launch_bounds__(256)
qkv_gemm_kernel(const unsigned short* __restrict__ xt,
                const unsigned short* __restrict__ Wt,
                const float* __restrict__ bias,
                unsigned short* __restrict__ out,
                int activation) {
    __shared__ unsigned short As[2][BM][LDA];
    __shared__ unsigned short Bs[2][BK][LDB2];
    const int tid  = threadIdx.x;
    const int wave = tid >> 5, lane = tid & 31;
    const int m0 = blockIdx.x * BM;
    const int n0 = blockIdx.y * BN;
    const int wm = (wave & 3) * 32;       // wave M offset
    const int wn = (wave >> 2) * 64;      // wave N offset

    const int ar = tid >> 2, ac = (tid & 3) * 8;    // A stage coords (128x32)
    const int br = tid >> 4, bc = (tid & 15) * 8;   // B stage coords (32x128)

    uint4 sa0, sa1, sb0, sb1;
    auto gload = [&](int kk) {
        sa0 = *reinterpret_cast<const uint4*>(xt + (size_t)(m0 + ar) * C_DIM + kk + ac);
        sa1 = *reinterpret_cast<const uint4*>(xt + (size_t)(m0 + ar + 64) * C_DIM + kk + ac);
        sb0 = *reinterpret_cast<const uint4*>(Wt + (size_t)(kk + br) * C_DIM + n0 + bc);
        sb1 = *reinterpret_cast<const uint4*>(Wt + (size_t)(kk + br + 16) * C_DIM + n0 + bc);
    };
    auto sstore = [&](int buf) {
        *reinterpret_cast<uint4*>(&As[buf][ar][ac])      = sa0;
        *reinterpret_cast<uint4*>(&As[buf][ar + 64][ac]) = sa1;
        *reinterpret_cast<uint4*>(&Bs[buf][br][bc])      = sb0;
        *reinterpret_cast<uint4*>(&Bs[buf][br + 16][bc]) = sb1;
    };

    gload(0);
    sstore(0);
    __syncthreads();

    floatx8 acc[2][4] = {};
    const int NT = C_DIM / BK;
    for (int kt = 0; kt < NT; ++kt) {
        const int buf = kt & 1;
        if (kt + 1 < NT) gload((kt + 1) * BK);   // issue next-tile loads early
        BFrag a0, a1;
        load_a_frag_t(As[buf], wm, lane, a0);
        load_a_frag_t(As[buf], wm + 16, lane, a1);
#pragma unroll
        for (int nb = 0; nb < 4; ++nb) {
            BFrag bf; load_b_frag_t(Bs[buf], wn + nb * 16, lane, bf);
            acc[0][nb] = wmma_bf16(a0, bf, acc[0][nb]);
            acc[1][nb] = wmma_bf16(a1, bf, acc[1][nb]);
        }
        if (kt + 1 < NT) sstore(buf ^ 1);
        __syncthreads();
    }

    const int rb = 8 * (lane >> 4), cl = lane & 15;
#pragma unroll
    for (int sub = 0; sub < 2; ++sub)
#pragma unroll
        for (int nb = 0; nb < 4; ++nb) {
            const int col = n0 + wn + nb * 16 + cl;
            const float bcol = bias[col];
#pragma unroll
            for (int r = 0; r < 8; ++r) {
                float v = acc[sub][nb][r] + bcol;
                if (activation) v = (v > 0.0f) ? v + 1.0f : __expf(v);  // elu+1
                out[(size_t)(m0 + wm + sub * 16 + rb + r) * C_DIM + col] = f2bf(v);
            }
        }
}

// ---- KV partials: KV_seg[b,h] = sum_{t in seg} k[t]^T v[t] ------------------
__global__ void __launch_bounds__(128)
kv_kernel(const unsigned short* __restrict__ k,
          const unsigned short* __restrict__ v,
          float* __restrict__ kvpart) {
    __shared__ unsigned short As[64][LDA];   // As[d][tt] = k[t][d] (transposed)
    __shared__ unsigned short Bs[32][LDB];   // Bs[tt][e] = v[t][e]
    const int tid  = threadIdx.x;
    const int wave = tid >> 5, lane = tid & 31;
    const int b = blockIdx.x / HEADS, h = blockIdx.x % HEADS;
    const int seg = blockIdx.y;
    const size_t base = (size_t)b * SEQ * C_DIM + (size_t)h * D_DIM;

    floatx8 acc[4] = {};
    for (int t0 = seg * TSEG; t0 < (seg + 1) * TSEG; t0 += 32) {
#pragma unroll
        for (int i = 0; i < 2; ++i) {
            const int chunk = tid + i * 128;          // 256 chunks of 8 shorts
            const int tt = chunk >> 3, d8 = (chunk & 7) * 8;
            union { uint4 u; unsigned short s[8]; } kd, vd;
            kd.u = *reinterpret_cast<const uint4*>(k + base + (size_t)(t0 + tt) * C_DIM + d8);
            vd.u = *reinterpret_cast<const uint4*>(v + base + (size_t)(t0 + tt) * C_DIM + d8);
            *reinterpret_cast<uint4*>(&Bs[tt][d8]) = vd.u;
#pragma unroll
            for (int j = 0; j < 8; ++j) As[d8 + j][tt] = kd.s[j];
        }
        __syncthreads();
        BFrag a; load_a_frag_t(As, wave * 16, lane, a);
#pragma unroll
        for (int nb = 0; nb < 4; ++nb) {
            BFrag bf; load_b_frag_t(Bs, nb * 16, lane, bf);
            acc[nb] = wmma_bf16(a, bf, acc[nb]);
        }
        __syncthreads();
    }
    float* dst = kvpart + (size_t)seg * (BATCH * HEADS * D_DIM * D_DIM)
               + (size_t)(b * HEADS + h) * D_DIM * D_DIM;
    const int rb = wave * 16 + 8 * (lane >> 4), cl = lane & 15;
#pragma unroll
    for (int nb = 0; nb < 4; ++nb)
#pragma unroll
        for (int r = 0; r < 8; ++r)
            dst[(rb + r) * D_DIM + nb * 16 + cl] = acc[nb][r];
}

__global__ void kv_reduce_kernel(const float* __restrict__ kvpart,
                                 unsigned short* __restrict__ kvb) {
    const int i = blockIdx.x * 256 + threadIdx.x;
    const int N = BATCH * HEADS * D_DIM * D_DIM;
    float s = 0.0f;
#pragma unroll
    for (int g = 0; g < KVSEG; ++g) s += kvpart[(size_t)g * N + i];
    kvb[i] = f2bf(s);
}

// ---- ksum partials & reduce (layout [B,H,D] == [B,C]) ----------------------
__global__ void ksum_part_kernel(const unsigned short* __restrict__ k,
                                 float* __restrict__ ksp) {
    const int idx = blockIdx.x * 256 + threadIdx.x;   // b*C + c
    const int seg = blockIdx.y;
    const int b = idx / C_DIM, c = idx % C_DIM;
    const unsigned short* kp = k + (size_t)b * SEQ * C_DIM + c;
    float s = 0.0f;
    for (int t = seg * TSEG; t < (seg + 1) * TSEG; ++t) s += bf2f(kp[(size_t)t * C_DIM]);
    ksp[(size_t)seg * (BATCH * C_DIM) + idx] = s;
}

__global__ void ksum_reduce_kernel(const float* __restrict__ ksp,
                                   float* __restrict__ ksum) {
    const int idx = blockIdx.x * 256 + threadIdx.x;
    float s = 0.0f;
#pragma unroll
    for (int g = 0; g < KVSEG; ++g) s += ksp[(size_t)g * (BATCH * C_DIM) + idx];
    ksum[idx] = s;
}

// ---- num = q @ KV ; y = num / (q.ksum + 1e-6) ------------------------------
__global__ void __launch_bounds__(128)
num_kernel(const unsigned short* __restrict__ q,
           const unsigned short* __restrict__ kvb,
           const float* __restrict__ ksum,
           unsigned short* __restrict__ y) {
    __shared__ unsigned short As[64][LDA];
    __shared__ unsigned short Bs[32][LDB];
    __shared__ float ks[64];
    __shared__ float zs[64];
    const int tid  = threadIdx.x;
    const int wave = tid >> 5, lane = tid & 31;
    const int t0 = blockIdx.x * 64;
    const int h  = blockIdx.y;
    const int b  = blockIdx.z;
    const size_t qbase  = (size_t)b * SEQ * C_DIM + (size_t)h * D_DIM;
    const size_t kvbase = (size_t)(b * HEADS + h) * D_DIM * D_DIM;

    if (tid < 64) ks[tid] = ksum[(b * HEADS + h) * D_DIM + tid];
    __syncthreads();
    if (tid < 64) {
        const unsigned short* qr = q + qbase + (size_t)(t0 + tid) * C_DIM;
        float z = 0.0f;
        for (int d = 0; d < D_DIM; ++d) z += bf2f(qr[d]) * ks[d];
        zs[tid] = z + 1e-6f;
    }

    floatx8 acc[4] = {};
    for (int kk = 0; kk < D_DIM; kk += 32) {
        for (int idx = tid; idx < 64 * 32; idx += 128) {
            int r = idx >> 5, c = idx & 31;
            As[r][c] = q[qbase + (size_t)(t0 + r) * C_DIM + kk + c];
        }
        for (int idx = tid; idx < 32 * 64; idx += 128) {
            int r = idx >> 6, c = idx & 63;
            Bs[r][c] = kvb[kvbase + (size_t)(kk + r) * D_DIM + c];
        }
        __syncthreads();
        BFrag a; load_a_frag_t(As, wave * 16, lane, a);
#pragma unroll
        for (int nb = 0; nb < 4; ++nb) {
            BFrag bf; load_b_frag_t(Bs, nb * 16, lane, bf);
            acc[nb] = wmma_bf16(a, bf, acc[nb]);
        }
        __syncthreads();
    }
    const int rbase = wave * 16 + 8 * (lane >> 4);
    const int col_l = lane & 15;
#pragma unroll
    for (int nb = 0; nb < 4; ++nb)
#pragma unroll
        for (int r = 0; r < 8; ++r) {
            const int lrow = rbase + r;
            y[qbase + (size_t)(t0 + lrow) * C_DIM + nb * 16 + col_l] =
                f2bf(acc[nb][r] / zs[lrow]);
        }
}

// ---- out = y @ Wo^T + bo + xt, stored transposed [B,C,T] -------------------
__global__ void __launch_bounds__(256)
out_gemm_kernel(const unsigned short* __restrict__ y,
                const unsigned short* __restrict__ Wot,
                const float* __restrict__ bo,
                const float* __restrict__ x,
                float* __restrict__ outp) {
    __shared__ unsigned short As[2][BM][LDA];
    __shared__ unsigned short Bs[2][BK][LDB2];
    __shared__ float Cs[64][LDC];
    const int tid  = threadIdx.x;
    const int wave = tid >> 5, lane = tid & 31;
    const int m0 = blockIdx.x * BM;
    const int n0 = blockIdx.y * BN;
    const int wm = (wave & 3) * 32;
    const int wn = (wave >> 2) * 64;

    const int ar = tid >> 2, ac = (tid & 3) * 8;
    const int br = tid >> 4, bc = (tid & 15) * 8;

    uint4 sa0, sa1, sb0, sb1;
    auto gload = [&](int kk) {
        sa0 = *reinterpret_cast<const uint4*>(y + (size_t)(m0 + ar) * C_DIM + kk + ac);
        sa1 = *reinterpret_cast<const uint4*>(y + (size_t)(m0 + ar + 64) * C_DIM + kk + ac);
        sb0 = *reinterpret_cast<const uint4*>(Wot + (size_t)(kk + br) * C_DIM + n0 + bc);
        sb1 = *reinterpret_cast<const uint4*>(Wot + (size_t)(kk + br + 16) * C_DIM + n0 + bc);
    };
    auto sstore = [&](int buf) {
        *reinterpret_cast<uint4*>(&As[buf][ar][ac])      = sa0;
        *reinterpret_cast<uint4*>(&As[buf][ar + 64][ac]) = sa1;
        *reinterpret_cast<uint4*>(&Bs[buf][br][bc])      = sb0;
        *reinterpret_cast<uint4*>(&Bs[buf][br + 16][bc]) = sb1;
    };

    gload(0);
    sstore(0);
    __syncthreads();

    floatx8 acc[2][4] = {};
    const int NT = C_DIM / BK;
    for (int kt = 0; kt < NT; ++kt) {
        const int buf = kt & 1;
        if (kt + 1 < NT) gload((kt + 1) * BK);
        BFrag a0, a1;
        load_a_frag_t(As[buf], wm, lane, a0);
        load_a_frag_t(As[buf], wm + 16, lane, a1);
#pragma unroll
        for (int nb = 0; nb < 4; ++nb) {
            BFrag bf; load_b_frag_t(Bs[buf], wn + nb * 16, lane, bf);
            acc[0][nb] = wmma_bf16(a0, bf, acc[0][nb]);
            acc[1][nb] = wmma_bf16(a1, bf, acc[1][nb]);
        }
        if (kt + 1 < NT) sstore(buf ^ 1);
        __syncthreads();
    }

    // Epilogue: two 64-row passes through LDS so [B,C,T] stores coalesce over T
    const int rb = 8 * (lane >> 4), cl = lane & 15;
    const int b  = m0 / SEQ;
    const int t0 = m0 % SEQ;
#pragma unroll
    for (int half = 0; half < 2; ++half) {
        if (((wave & 3) >> 1) == half) {
            const int lr = ((wave & 3) & 1) * 32;
#pragma unroll
            for (int sub = 0; sub < 2; ++sub)
#pragma unroll
                for (int nb = 0; nb < 4; ++nb)
#pragma unroll
                    for (int r = 0; r < 8; ++r)
                        Cs[lr + sub * 16 + rb + r][wn + nb * 16 + cl] = acc[sub][nb][r];
        }
        __syncthreads();
        for (int idx = tid; idx < 64 * 128; idx += 256) {
            const int tl = idx & 63, c = idx >> 6;   // adjacent threads -> adjacent t
            const int cg = n0 + c;
            const size_t gi = ((size_t)b * C_DIM + cg) * SEQ + t0 + half * 64 + tl;
            outp[gi] = Cs[tl][c] + bo[cg] + x[gi];
        }
        __syncthreads();
    }
}

extern "C" void kernel_launch(void* const* d_in, const int* in_sizes, int n_in,
                              void* d_out, int out_size, void* d_ws, size_t ws_size,
                              hipStream_t stream) {
    (void)in_sizes; (void)n_in; (void)out_size; (void)ws_size;
    const float* x  = (const float*)d_in[0];
    const float* Wq = (const float*)d_in[1];
    const float* bq = (const float*)d_in[2];
    const float* Wk = (const float*)d_in[3];
    const float* bk = (const float*)d_in[4];
    const float* Wv = (const float*)d_in[5];
    const float* bv = (const float*)d_in[6];
    const float* Wo = (const float*)d_in[7];
    const float* bo = (const float*)d_in[8];
    float* outp = (float*)d_out;

    char* ws = (char*)d_ws;
    size_t off = 0;
    auto alloc = [&](size_t bytes) -> void* {
        void* p = ws + off;
        off += (bytes + 255) & ~(size_t)255;
        return p;
    };
    unsigned short* xt  = (unsigned short*)alloc((size_t)BT * C_DIM * 2);
    unsigned short* Wqt = (unsigned short*)alloc((size_t)C_DIM * C_DIM * 2);
    unsigned short* Wkt = (unsigned short*)alloc((size_t)C_DIM * C_DIM * 2);
    unsigned short* Wvt = (unsigned short*)alloc((size_t)C_DIM * C_DIM * 2);
    unsigned short* Wot = (unsigned short*)alloc((size_t)C_DIM * C_DIM * 2);
    unsigned short* qb  = (unsigned short*)alloc((size_t)BT * C_DIM * 2);
    unsigned short* kb  = (unsigned short*)alloc((size_t)BT * C_DIM * 2);
    unsigned short* vb  = (unsigned short*)alloc((size_t)BT * C_DIM * 2);
    unsigned short* kvb = (unsigned short*)alloc((size_t)BATCH * HEADS * D_DIM * D_DIM * 2);
    float*          kvp = (float*)alloc((size_t)KVSEG * BATCH * HEADS * D_DIM * D_DIM * 4);
    float*          ksp = (float*)alloc((size_t)KVSEG * BATCH * C_DIM * 4);
    float*          ks  = (float*)alloc((size_t)BATCH * C_DIM * 4);
    unsigned short* yb  = (unsigned short*)alloc((size_t)BT * C_DIM * 2);

    const dim3 tblk(32, 8);
    transpose_x_kernel<<<dim3(SEQ / 32, C_DIM / 32, BATCH), tblk, 0, stream>>>(x, xt);
    transpose_w_kernel<<<dim3(C_DIM / 32, C_DIM / 32), tblk, 0, stream>>>(Wq, Wqt);
    transpose_w_kernel<<<dim3(C_DIM / 32, C_DIM / 32), tblk, 0, stream>>>(Wk, Wkt);
    transpose_w_kernel<<<dim3(C_DIM / 32, C_DIM / 32), tblk, 0, stream>>>(Wv, Wvt);
    transpose_w_kernel<<<dim3(C_DIM / 32, C_DIM / 32), tblk, 0, stream>>>(Wo, Wot);

    const dim3 ggrid(BT / BM, C_DIM / BN);
    qkv_gemm_kernel<<<ggrid, 256, 0, stream>>>(xt, Wqt, bq, qb, 1);
    qkv_gemm_kernel<<<ggrid, 256, 0, stream>>>(xt, Wkt, bk, kb, 1);
    qkv_gemm_kernel<<<ggrid, 256, 0, stream>>>(xt, Wvt, bv, vb, 0);

    ksum_part_kernel<<<dim3((BATCH * C_DIM) / 256, KVSEG), 256, 0, stream>>>(kb, ksp);
    ksum_reduce_kernel<<<(BATCH * C_DIM) / 256, 256, 0, stream>>>(ksp, ks);

    kv_kernel<<<dim3(BATCH * HEADS, KVSEG), 128, 0, stream>>>(kb, vb, kvp);
    kv_reduce_kernel<<<(BATCH * HEADS * D_DIM * D_DIM) / 256, 256, 0, stream>>>(kvp, kvb);

    num_kernel<<<dim3(SEQ / 64, HEADS, BATCH), 128, 0, stream>>>(qb, kvb, ks, yb);
    out_gemm_kernel<<<ggrid, 256, 0, stream>>>(yb, Wot, bo, x, outp);
}
